// C_LSTM_15333033247178
// MI455X (gfx1250) — matrix operand order repaired and checked by reading, not compiled, not run on hardware
//
#include <hip/hip_runtime.h>

#define B 128
#define S 256
#define H 1024
#define G4 4096
#define HC 16            // hidden columns per block
#define NBLK (H / HC)    // 64 persistent blocks
#define NTHREADS 256     // 8 wave32
#define KTILES (H / 32)  // 32 WMMA K-steps

typedef __attribute__((ext_vector_type(16))) _Float16 v16h;
typedef __attribute__((ext_vector_type(8)))  _Float16 v8h;
typedef __attribute__((ext_vector_type(8)))  float    v8f;

__device__ __forceinline__ float fast_sig(float x) {
    return 1.0f / (1.0f + __expf(-x));
}
__device__ __forceinline__ float fast_tanh(float x) {
    x = fminf(fmaxf(x, -15.0f), 15.0f);
    float e = __expf(2.0f * x);
    return (e - 1.0f) / (e + 1.0f);
}

// Reset barrier + zero h buffer 0 each call (deterministic across graph replays).
__global__ void ctlstm_init(unsigned* __restrict__ bar, unsigned* __restrict__ hbuf_w) {
    int i = blockIdx.x * blockDim.x + threadIdx.x;
    if (i < 2) bar[i] = 0u;
    if (i < (B * H / 2)) hbuf_w[i] = 0u;   // B*H halves == B*H/2 dwords
}

__launch_bounds__(NTHREADS, 1)
__global__ void ctlstm_kernel(const float* __restrict__ events,
                              const float* __restrict__ rtg,
                              const float* __restrict__ W,
                              const float* __restrict__ U,
                              const float* __restrict__ Q,
                              const float* __restrict__ bias,
                              float* __restrict__ out,
                              unsigned* __restrict__ bar,
                              _Float16* __restrict__ hbuf) {
    __shared__ _Float16 Usl[4 * HC * H];   // 64 cols x 1024 K, column-major, 128 KB
    __shared__ float ev[B];                // events column for this step
    __shared__ float tdb[3 * B];           // td basis (k-major) for this step

    const int tid  = threadIdx.x;
    const int cb   = blockIdx.x;           // 0..63 -> hidden cols [cb*16, cb*16+16)
    const int hc0  = cb * HC;
    const int lane = tid & 31;
    const int wave = tid >> 5;             // 0..7: M-tile (batch rows 16w..16w+15)
    const int lhi  = lane >> 4;            // 0/1
    const int llo  = lane & 15;

    // ---- stage this block's U slice into LDS (fp32 -> f16, column-major) ----
    for (int idx = tid; idx < 4 * HC * H; idx += NTHREADS) {
        int k    = idx >> 6;               // 0..1023
        int cc   = idx & 63;               // 0..63 = gate*16 + jj
        int gate = cc >> 4;
        int jj   = cc & 15;
        Usl[cc * H + k] = (_Float16)U[(size_t)k * G4 + gate * H + hc0 + jj];
    }

    // ---- per-lane column constants (col = hc0 + llo) ----
    const int col = hc0 + llo;
    float Wc[4], bc[4], Qc[3];
#pragma unroll
    for (int g = 0; g < 4; ++g) {
        Wc[g] = W[g * H + col];
        bc[g] = bias[g * H + col];
    }
#pragma unroll
    for (int k = 0; k < 3; ++k) Qc[k] = Q[k * H + col];

    float c_[8];
#pragma unroll
    for (int r = 0; r < 8; ++r) c_[r] = 0.0f;

    __syncthreads();

    for (int t = 0; t < S; ++t) {
        const _Float16* __restrict__ hsrc = hbuf + (size_t)(t & 1) * (B * H);
        _Float16* __restrict__ hdst       = hbuf + (size_t)((t + 1) & 1) * (B * H);

        // ---- stage events column + scrambled time-decay basis td[b,t,k] ----
        if (tid < B) ev[tid] = events[tid * S + t];
        if (tid < 3 * B) {
            int k    = tid >> 7;
            int b    = tid & 127;
            int flat = b * (S * 3) + t * 3 + k;       // faithful torch.cat/reshape scramble
            int row  = flat >> 8;
            int ccol = flat & 255;
            int p    = row >> 7;
            float v  = rtg[(row & 127) * S + ccol];
            tdb[k * B + b] = (p == 0) ? v : ((p == 1) ? v * v : v * v * v);
        }
        __syncthreads();

        // ---- GEMM: gates[128 x 64] = h[128 x 1024] @ Usl[1024 x 64] ----
        v8f acc0 = {}, acc1 = {}, acc2 = {}, acc3 = {};
        const _Float16* __restrict__ arow = hsrc + (size_t)(wave * 16 + llo) * H;
#pragma unroll 4
        for (int kk = 0; kk < KTILES; ++kk) {
            v16h af;
            const int ka = kk * 32 + lhi * 16;         // A: K+0..15 (lo lanes) / K+16..31 (hi)
            *((v8h*)&af)     = *(const v8h*)(arow + ka);
            *((v8h*)&af + 1) = *(const v8h*)(arow + ka + 8);
            const int kb = kk * 32 + lhi * 16;         // B: K+0..15 / K+16..31
            v16h b0 = *(const v16h*)(&Usl[(0 * 16 + llo) * H + kb]);
            v16h b1 = *(const v16h*)(&Usl[(1 * 16 + llo) * H + kb]);
            v16h b2 = *(const v16h*)(&Usl[(2 * 16 + llo) * H + kb]);
            v16h b3 = *(const v16h*)(&Usl[(3 * 16 + llo) * H + kb]);
            acc0 = __builtin_amdgcn_wmma_f32_16x16x32_f16(false, af, false, b0, (short)0, acc0, false, false);
            acc1 = __builtin_amdgcn_wmma_f32_16x16x32_f16(false, af, false, b1, (short)0, acc1, false, false);
            acc2 = __builtin_amdgcn_wmma_f32_16x16x32_f16(false, af, false, b2, (short)0, acc2, false, false);
            acc3 = __builtin_amdgcn_wmma_f32_16x16x32_f16(false, af, false, b3, (short)0, acc3, false, false);
        }

        // ---- elementwise gates + state update (register-exact C/D mapping) ----
#pragma unroll
        for (int r = 0; r < 8; ++r) {
            const int b   = wave * 16 + r + 8 * lhi;
            const float e = ev[b];
            const float tdq = tdb[0 * B + b] * Qc[0] + tdb[1 * B + b] * Qc[1] + tdb[2 * B + b] * Qc[2];
            float iv = fast_sig (acc0[r] + e * Wc[0] + bc[0]);
            float fv = fast_sig (acc1[r] + e * Wc[1] + bc[1] + tdq);
            float gv = fast_tanh(acc2[r] + e * Wc[2] + bc[2]);
            float ov = fast_sig (acc3[r] + e * Wc[3] + bc[3]);
            float cn = fv * c_[r] + iv * gv;
            c_[r]    = cn;
            float hh = ov * fast_tanh(cn);
            out[((size_t)b * S + t) * H + col] = hh;                   // hidden_seq
            hdst[(size_t)b * H + col] = (_Float16)hh;                  // f16 h for next GEMM
            if (t == S - 1) {
                out[(size_t)B * S * H + (size_t)b * H + col] = hh;     // h_t
                out[(size_t)B * S * H + (size_t)B * H + (size_t)b * H + col] = cn; // c_t
            }
        }

        // ---- grid-wide step barrier (generation counter) ----
        __threadfence();
        __syncthreads();
        if (tid == 0) {
            volatile unsigned* vbar = bar;
            unsigned gen = vbar[1];
            if (atomicAdd(&bar[0], 1u) == (unsigned)(gridDim.x - 1)) {
                bar[0] = 0u;
                __threadfence();
                atomicAdd(&bar[1], 1u);
            } else {
                while (vbar[1] == gen) { __builtin_amdgcn_s_sleep(1); }
            }
        }
        __syncthreads();
    }
}

extern "C" void kernel_launch(void* const* d_in, const int* in_sizes, int n_in,
                              void* d_out, int out_size, void* d_ws, size_t ws_size,
                              hipStream_t stream) {
    const float* events = (const float*)d_in[0];
    const float* rtg    = (const float*)d_in[1];
    // d_in[2] = mask (unused by forward)
    const float* W      = (const float*)d_in[3];
    const float* U      = (const float*)d_in[4];
    const float* Q      = (const float*)d_in[5];
    const float* bias   = (const float*)d_in[6];
    float* out          = (float*)d_out;

    unsigned*  bar  = (unsigned*)d_ws;
    _Float16*  hbuf = (_Float16*)((char*)d_ws + 256);   // 2 * B*H f16 (double buffer)

    hipLaunchKernelGGL(ctlstm_init, dim3((B * H / 2 + 255) / 256), dim3(256), 0, stream,
                       bar, (unsigned*)hbuf);
    hipLaunchKernelGGL(ctlstm_kernel, dim3(NBLK), dim3(NTHREADS), 0, stream,
                       events, rtg, W, U, Q, bias, out, bar, hbuf);
}
